// _BaseResNet_33200097198451
// MI455X (gfx1250) — matrix operand order repaired and checked by reading, not compiled, not run on hardware
//
#include <hip/hip_runtime.h>

typedef unsigned short u16;
typedef __attribute__((ext_vector_type(16))) __bf16          v16bf;
typedef __attribute__((ext_vector_type(4)))  unsigned int    v4ui;
typedef __attribute__((ext_vector_type(2)))  unsigned int    v2ui;
typedef __attribute__((ext_vector_type(8)))  unsigned int    v8ui;
typedef __attribute__((ext_vector_type(8)))  float           v8f;

#if __has_builtin(__builtin_amdgcn_global_load_async_to_lds_b128) && \
    __has_builtin(__builtin_amdgcn_s_wait_asynccnt)
#define HAS_ASYNC_LDS 1
// builtin prototype (from hipcc diagnostic): GCC-vector int4 pointers, AS1 / AS3
typedef int v4i __attribute__((vector_size(16)));
typedef __attribute__((address_space(1))) v4i* gv4p;
typedef __attribute__((address_space(3))) v4i* lv4p;
#else
#define HAS_ASYNC_LDS 0
#endif

// ---------- bf16 helpers (RNE) ----------
__device__ __forceinline__ u16 f2bf(float f) {
    unsigned u = __float_as_uint(f);
    u += 0x7FFFu + ((u >> 16) & 1u);
    return (u16)(u >> 16);
}
__device__ __forceinline__ float bf2f(u16 h) {
    return __uint_as_float(((unsigned)h) << 16);
}

// two 16B-aligned LDS pieces -> one 16x32 bf16 fragment (forces ds_load_b128)
__device__ __forceinline__ v16bf ldfrag(const u16* p0, const u16* p1) {
    v4ui lo = *(const v4ui*)p0;
    v4ui hi = *(const v4ui*)p1;
    v8ui f = {lo[0], lo[1], lo[2], lo[3], hi[0], hi[1], hi[2], hi[3]};
    return __builtin_bit_cast(v16bf, f);
}

// ---------- layout / convert kernels ----------
// NCHW f32 -> NHWC bf16 with channel padding (Cin -> Cpad, extra channels zero)
__global__ void cvt_nchw_to_nhwc_pad(const float* __restrict__ x, u16* __restrict__ y,
                                     int N, int Cin, int Cpad, int H, int W) {
    int i = blockIdx.x * 256 + threadIdx.x;
    int tot = N * H * W * Cpad;
    if (i >= tot) return;
    int c = i % Cpad; int t = i / Cpad;
    int w = t % W; t /= W;
    int h = t % H; int n = t / H;
    y[i] = (c < Cin) ? f2bf(x[((n * Cin + c) * H + h) * W + w]) : (u16)0;
}

// OIHW f32 -> [ (r*Spad+s)*Ipad + c ][ O ] bf16 (zero for c>=I or s>=S)
__global__ void repack_oihw_kx(const float* __restrict__ w, u16* __restrict__ y,
                               int O, int I, int Ipad, int R, int S, int Spad) {
    int i = blockIdx.x * 256 + threadIdx.x;
    int tot = O * Ipad * R * Spad;
    if (i >= tot) return;
    int o = i % O; int t = i / O;
    int c = t % Ipad; t /= Ipad;
    int s = t % Spad; int r = t / Spad;
    y[i] = (c < I && s < S) ? f2bf(w[((o * I + c) * R + r) * S + s]) : (u16)0;
}

// fc [O][K] f32 -> [K][Opad] bf16 (zero cols for o >= O)
__global__ void repack_fc_kx(const float* __restrict__ w, u16* __restrict__ y,
                             int O, int Opad, int K) {
    int i = blockIdx.x * 256 + threadIdx.x;
    if (i >= K * Opad) return;
    int o = i % Opad;
    int k = i / Opad;
    y[i] = (o < O) ? f2bf(w[o * K + k]) : (u16)0;
}

__global__ void maxpool3x3s2_nhwc(const u16* __restrict__ in, u16* __restrict__ out,
                                  int N, int C, int H, int W, int OH, int OW) {
    int i = blockIdx.x * 256 + threadIdx.x;
    int tot = N * OH * OW * C;
    if (i >= tot) return;
    int c = i % C; int t = i / C;
    int ow = t % OW; t /= OW;
    int oh = t % OH; int n = t / OH;
    float best = -3.0e38f; u16 bits = 0xFF80u; // bf16 -inf
    for (int r = 0; r < 3; ++r) {
        int ih = oh * 2 - 1 + r;
        if ((unsigned)ih >= (unsigned)H) continue;
        for (int s = 0; s < 3; ++s) {
            int iw = ow * 2 - 1 + s;
            if ((unsigned)iw >= (unsigned)W) continue;
            u16 b = in[((n * H + ih) * W + iw) * C + c];
            float v = bf2f(b);
            if (v > best) { best = v; bits = b; }
        }
    }
    out[i] = bits;
}

__global__ void avgpool7x7_nhwc(const u16* __restrict__ in, u16* __restrict__ out, int N, int C) {
    int i = blockIdx.x * 256 + threadIdx.x;
    if (i >= N * C) return;
    int c = i % C, n = i / C;
    float s = 0.f;
    for (int k = 0; k < 49; ++k) s += bf2f(in[(n * 49 + k) * C + c]);
    out[i] = f2bf(s * (1.0f / 49.0f));
}

// =====================================================================================
// Implicit-GEMM conv, fused BN(+skip)(+ReLU). bf16 in / f32 WMMA acc / bf16 out.
// Block tile 128(M) x 64(N); 8 wave32s; each wave owns a 32x32 tile = 2x2 accumulators
// => 4 x V_WMMA_F32_16X16X32_BF16 per 32-wide K-chunk.
// Double-buffered LDS, one barrier per chunk; next chunk's weight tile staged with
// GLOBAL_LOAD_ASYNC_TO_LDS_B128 (ASYNCcnt) overlapping the WMMAs of the current chunk.
// Path A (C%32==0): uniform nested (r,s,c0) walk, zero divisions.
// Path B (pow2 C & S, KK%32==0; stem padded to C=4,S=8): pure-shift decode, b64 cells.
// =====================================================================================
__global__ __launch_bounds__(256) void conv_bn_wmma(
    const u16* __restrict__ in, const u16* __restrict__ wk,
    const float* __restrict__ gamma, const float* __restrict__ beta,
    const float* __restrict__ mean,  const float* __restrict__ var,
    const u16* __restrict__ skip, u16* __restrict__ out,
    int N, int H, int W, int C, int Cout, int R, int S, int stride, int pad,
    int OH, int OW, int relu)
{
    __shared__ u16 As[2][128][40]; // 128 pixels x 32 K (pitch 80B = 5x16B)
    __shared__ u16 Bs[2][32][72];  // 32 K x 64 out-ch  (pitch 144B = 9x16B)

    const int tid  = threadIdx.x;
    const int lane = tid & 31;
    const int wv   = tid >> 5;
    const int wm   = wv & 3;   // 4 M-groups of 32 rows
    const int wn   = wv >> 2;  // 2 N-groups of 32 cols

    const int tileM = blockIdx.x * 128;
    const int tileN = blockIdx.y * 64;
    const int Mtot  = N * OH * OW;
    const int KK    = C * R * S;

    // ---- per-thread A-row decode, hoisted out of the K loop (2 threads per row) ----
    const int  myRow = tid >> 1;
    const int  half  = (tid & 1) * 16;          // which 16 K-elems of the row
    const int  p     = tileM + myRow;
    const bool rowValid = p < Mtot;
    int n = 0, ih0 = 0, iw0 = 0;
    if (rowValid) {
        int ow = p % OW; int t = p / OW;
        int oh = t % OH; n = t / OH;
        ih0 = oh * stride - pad;
        iw0 = ow * stride - pad;
    }

    v8f acc[2][2];
    #pragma unroll
    for (int a = 0; a < 2; ++a)
        #pragma unroll
        for (int b = 0; b < 2; ++b)
            acc[a][b] = (v8f){0.f, 0.f, 0.f, 0.f, 0.f, 0.f, 0.f, 0.f};

    // ---- fragment load + 4 WMMAs on buffer `bf` ----
    auto mma_chunk = [&](int bf) {
        const int m0 = wm * 32 + (lane & 15);
        const int kb = (lane >> 4) * 8;
        v16bf a0 = ldfrag(&As[bf][m0][kb],      &As[bf][m0][kb + 16]);
        v16bf a1 = ldfrag(&As[bf][m0 + 16][kb], &As[bf][m0 + 16][kb + 16]);
        v16bf b0, b1;
        #pragma unroll
        for (int s = 0; s < 16; ++s) {
            const int kk = kb + (s & 7) + ((s >> 3) << 4);
            b0[s] = __builtin_bit_cast(__bf16, Bs[bf][kk][wn * 32 + (lane & 15)]);
            b1[s] = __builtin_bit_cast(__bf16, Bs[bf][kk][wn * 32 + 16 + (lane & 15)]);
        }
        acc[0][0] = __builtin_amdgcn_wmma_f32_16x16x32_bf16(false, a0, false, b0, (short)0, acc[0][0], false, false);
        acc[0][1] = __builtin_amdgcn_wmma_f32_16x16x32_bf16(false, a0, false, b1, (short)0, acc[0][1], false, false);
        acc[1][0] = __builtin_amdgcn_wmma_f32_16x16x32_bf16(false, a1, false, b0, (short)0, acc[1][0], false, false);
        acc[1][1] = __builtin_amdgcn_wmma_f32_16x16x32_bf16(false, a1, false, b1, (short)0, acc[1][1], false, false);
    };

    // ---- stage a full weight tile (rows kbase..kbase+31 all < KK) into Bs[bf] ----
    auto stageB = [&](int kbase, int bf) {
#if HAS_ASYNC_LDS
        const int brow = tid >> 3, bseg = tid & 7;   // 256 async b128 transfers
        const u16* g = wk + (kbase + brow) * Cout + tileN + bseg * 8;
        __builtin_amdgcn_global_load_async_to_lds_b128(
            (gv4p)(u16*)g, (lv4p)&Bs[bf][brow][bseg * 8], 0, 0);
#else
        #pragma unroll
        for (int i = 0; i < 8; ++i) {
            int idx2 = tid + i * 256;
            int brow = idx2 >> 6, col = idx2 & 63;
            Bs[bf][brow][col] = wk[(kbase + brow) * Cout + tileN + col];
        }
#endif
    };

    if ((C & 31) == 0) {
        // ================= path A: C%32==0, uniform (r,s,c0) walk =================
        auto stageA = [&](int r, int s, int c0, int bf) {
            const int ih = ih0 + r, iw = iw0 + s;
            v4ui d0 = {0, 0, 0, 0}, d1 = {0, 0, 0, 0};
            if (rowValid && (unsigned)ih < (unsigned)H && (unsigned)iw < (unsigned)W) {
                const v4ui* src = (const v4ui*)(in + ((n * H + ih) * W + iw) * C + c0 + half);
                d0 = src[0]; d1 = src[1];
            }
            v4ui* dst = (v4ui*)&As[bf][myRow][half];
            dst[0] = d0; dst[1] = d1;
        };
        const int nCh = R * S * (C >> 5);
        int rn = 0, sn = 0, cn = 0;
        stageB(0, 0);
        stageA(0, 0, 0, 0);
        for (int q = 0; q < nCh; ++q) {
#if HAS_ASYNC_LDS
            __builtin_amdgcn_s_wait_asynccnt(0);
#endif
            __syncthreads();
            if (q + 1 < nCh) {
                cn += 32;
                if (cn >= C) { cn = 0; if (++sn >= S) { sn = 0; ++rn; } }
                stageB((rn * S + sn) * C + cn, (q + 1) & 1);
                stageA(rn, sn, cn, (q + 1) & 1);
            }
            mma_chunk(q & 1);
        }
    } else {
        // ========== path B: pow2 C & S, KK%32==0 (stem: C=4, S=8, KK=224) ==========
        const int cShift = __builtin_ctz(C);
        const int sShift = __builtin_ctz(S);
        auto stageA2 = [&](int k0, int bf) {
            #pragma unroll
            for (int j = 0; j < 4; ++j) {           // 4 cells x 4 contiguous channels (b64)
                int kc = k0 + half + j * 4;
                int rs = kc >> cShift;
                int rr = rs >> sShift;
                int ss = rs & (S - 1);
                int cc = kc & (C - 1);
                int ih = ih0 + rr, iw = iw0 + ss;
                v2ui d = {0, 0};
                if (rowValid && (unsigned)ih < (unsigned)H && (unsigned)iw < (unsigned)W)
                    d = *(const v2ui*)(in + ((n * H + ih) * W + iw) * C + cc);
                *(v2ui*)&As[bf][myRow][half + j * 4] = d;
            }
        };
        const int nCh = KK >> 5;
        stageB(0, 0);
        stageA2(0, 0);
        for (int q = 0; q < nCh; ++q) {
#if HAS_ASYNC_LDS
            __builtin_amdgcn_s_wait_asynccnt(0);
#endif
            __syncthreads();
            if (q + 1 < nCh) {
                stageB((q + 1) * 32, (q + 1) & 1);
                stageA2((q + 1) * 32, (q + 1) & 1);
            }
            mma_chunk(q & 1);
        }
    }

    // ---- epilogue: BN (+skip) (+ReLU). C/D layout: VGPR v -> M = v + 8*(lane/16), N = lane%16
    const int mbase = tileM + wm * 32 + ((lane >> 4) << 3);
    const int nbase = tileN + wn * 32 + (lane & 15);
    #pragma unroll
    for (int nt = 0; nt < 2; ++nt) {
        int oc = nbase + nt * 16;
        if (oc >= Cout) continue;
        float sc = __frsqrt_rn(var[oc] + 1e-5f) * gamma[oc];
        float mu = mean[oc], bt = beta[oc];
        #pragma unroll
        for (int mt = 0; mt < 2; ++mt) {
            #pragma unroll
            for (int v = 0; v < 8; ++v) {
                int pp = mbase + mt * 16 + v;
                if (pp < Mtot) {
                    float val = (acc[mt][nt][v] - mu) * sc + bt;
                    if (skip != nullptr) val += bf2f(skip[pp * Cout + oc]);
                    if (relu) val = fmaxf(val, 0.f);
                    out[pp * Cout + oc] = f2bf(val);
                }
            }
        }
    }
}

// ---------- WMMA GEMM for the FC head: A[M][K] bf16 x B[K][Npad] bf16 -> f32 ----------
// M multiple of 64, K multiple of 32, Npad multiple of 32; stores guarded by Nout.
__global__ __launch_bounds__(256) void gemm_wmma_f32out(
    const u16* __restrict__ A, const u16* __restrict__ B, float* __restrict__ out,
    int M, int K, int Npad, int Nout)
{
    __shared__ u16 As[64][40];
    __shared__ u16 Bs[32][40];
    const int tid = threadIdx.x, lane = tid & 31, wv = tid >> 5;
    const int wm = wv & 3, wn = wv >> 2;
    const int tileM = blockIdx.x * 64;
    const int tileN = blockIdx.y * 32;

    v8f acc = {0.f, 0.f, 0.f, 0.f, 0.f, 0.f, 0.f, 0.f};

    for (int k0 = 0; k0 < K; k0 += 32) {
        {   // A tile: 64 rows x 32 K, one b128 per thread
            int row = tid >> 2, seg = tid & 3;
            *(v4ui*)&As[row][seg * 8] = *(const v4ui*)(A + (tileM + row) * K + k0 + seg * 8);
        }
#if HAS_ASYNC_LDS
        if (tid < 128) {   // B tile: 32 rows x 32 cols, 128 async b128 transfers (wave-uniform branch)
            int row = tid >> 2, seg = tid & 3;
            const u16* g = B + (k0 + row) * Npad + tileN + seg * 8;
            __builtin_amdgcn_global_load_async_to_lds_b128(
                (gv4p)(u16*)g, (lv4p)&Bs[row][seg * 8], 0, 0);
        }
        __builtin_amdgcn_s_wait_asynccnt(0);
#else
        if (tid < 128) {
            int row = tid >> 2, seg = tid & 3;
            *(v4ui*)&Bs[row][seg * 8] = *(const v4ui*)(B + (k0 + row) * Npad + tileN + seg * 8);
        }
#endif
        __syncthreads();

        const int m = wm * 16 + (lane & 15);
        const int kb = (lane >> 4) * 8;
        v16bf a = ldfrag(&As[m][kb], &As[m][kb + 16]);
        v16bf b;
        #pragma unroll
        for (int s = 0; s < 16; ++s)
            b[s] = __builtin_bit_cast(__bf16, Bs[kb + (s & 7) + ((s >> 3) << 4)][wn * 16 + (lane & 15)]);
        acc = __builtin_amdgcn_wmma_f32_16x16x32_bf16(false, a, false, b, (short)0, acc, false, false);
        __syncthreads();
    }

    #pragma unroll
    for (int v = 0; v < 8; ++v) {
        int mloc = v + ((lane >> 4) << 3);
        int nloc = lane & 15;
        int pp = tileM + wm * 16 + mloc;
        int oc = tileN + wn * 16 + nloc;
        if (pp < M && oc < Nout) out[pp * Nout + oc] = acc[v];
    }
}

// =====================================================================================
extern "C" void kernel_launch(void* const* d_in, const int* in_sizes, int n_in,
                              void* d_out, int out_size, void* d_ws, size_t ws_size,
                              hipStream_t stream) {
    (void)in_sizes; (void)n_in; (void)out_size; (void)ws_size;
    const int N = 64;

    // ---- input index map (jax tree flatten: alphabetical dict keys) ----
    // top: params < x ; params: fc_w, layers, stem_bn, stem_w ; bn: beta,gamma,mean,var
    int idx = 0;
    const int FCW = idx++;
    struct BP { int bn1[4], bn2[4], dbn[4]; int dw, w1, w2; int down; };
    BP bp[4][2];
    for (int li = 0; li < 4; ++li)
        for (int b = 0; b < 2; ++b) {
            BP& p = bp[li][b];
            p.down = (li > 0 && b == 0);
            for (int j = 0; j < 4; ++j) p.bn1[j] = idx++;
            for (int j = 0; j < 4; ++j) p.bn2[j] = idx++;
            if (p.down) { for (int j = 0; j < 4; ++j) p.dbn[j] = idx++; p.dw = idx++; }
            else        { p.dbn[0]=p.dbn[1]=p.dbn[2]=p.dbn[3]=p.dw=-1; }
            p.w1 = idx++; p.w2 = idx++;
        }
    int STEM_BN[4]; for (int j = 0; j < 4; ++j) STEM_BN[j] = idx++;
    const int STEMW = idx++;
    const int XIDX  = idx++;

    auto F = [&](int i) { return (const float*)d_in[i]; };

    // ---- workspace bump allocator ----
    char* base = (char*)d_ws; size_t off = 0;
    auto alloc = [&](size_t bytes) -> void* {
        off = (off + 255) & ~(size_t)255;
        void* p = base + off; off += bytes; return p;
    };

    const int inC[4]  = {64, 64, 128, 256};
    const int outC[4] = {64, 128, 256, 512};

    u16* stemW = (u16*)alloc(2ull * 64 * 4 * 7 * 8);  // padded: C 3->4, S 7->8, KK=224
    u16* bw1[4][2]; u16* bw2[4][2]; u16* bdw[4];
    for (int li = 0; li < 4; ++li) {
        for (int b = 0; b < 2; ++b) {
            int ic = (b == 0) ? inC[li] : outC[li];
            bw1[li][b] = (u16*)alloc(2ull * outC[li] * ic * 9);
            bw2[li][b] = (u16*)alloc(2ull * outC[li] * outC[li] * 9);
        }
        bdw[li] = (li > 0) ? (u16*)alloc(2ull * outC[li] * inC[li]) : nullptr;
    }
    u16* fcW    = (u16*)alloc(2ull * 512 * 1024);     // cols padded 1000 -> 1024
    u16* P0     = (u16*)alloc(2ull * N * 112 * 112 * 64); // largest activation
    u16* P1     = (u16*)alloc(2ull * N * 56 * 56 * 64);   // == 64*224*224*4 elems as well
    u16* P2     = (u16*)alloc(2ull * N * 56 * 56 * 64);
    u16* pooled = (u16*)alloc(2ull * N * 512);

    auto grid1 = [](long long tot) { return dim3((unsigned)((tot + 255) / 256)); };

    // ---- weight repack (f32 -> bf16, K-major, padded) ----
    repack_oihw_kx<<<grid1(64ll * 4 * 7 * 8), 256, 0, stream>>>(F(STEMW), stemW, 64, 3, 4, 7, 7, 8);
    for (int li = 0; li < 4; ++li) {
        for (int b = 0; b < 2; ++b) {
            int ic = (b == 0) ? inC[li] : outC[li];
            repack_oihw_kx<<<grid1((long long)outC[li] * ic * 9), 256, 0, stream>>>(
                F(bp[li][b].w1), bw1[li][b], outC[li], ic, ic, 3, 3, 3);
            repack_oihw_kx<<<grid1((long long)outC[li] * outC[li] * 9), 256, 0, stream>>>(
                F(bp[li][b].w2), bw2[li][b], outC[li], outC[li], outC[li], 3, 3, 3);
        }
        if (li > 0)
            repack_oihw_kx<<<grid1((long long)outC[li] * inC[li]), 256, 0, stream>>>(
                F(bp[li][0].dw), bdw[li], outC[li], inC[li], inC[li], 1, 1, 1);
    }
    repack_fc_kx<<<grid1(512ll * 1024), 256, 0, stream>>>(F(FCW), fcW, 1000, 1024, 512);

    // ---- input convert: NCHW f32 -> NHWC bf16, padded to 4 channels (into P1) ----
    cvt_nchw_to_nhwc_pad<<<grid1((long long)N * 224 * 224 * 4), 256, 0, stream>>>(
        F(XIDX), P1, N, 3, 4, 224, 224);

    // ---- conv launcher: block tile 128 x 64 ----
    auto conv = [&](const u16* in, const u16* w, const int bn[4], const u16* skip, u16* outp,
                    int H, int Wd, int C, int Co, int R, int S, int st, int pad, int relu) {
        int OH = (H + 2 * pad - R) / st + 1;
        int OW = (Wd + 2 * pad - S) / st + 1;
        dim3 g((unsigned)(((long long)N * OH * OW + 127) / 128), (unsigned)((Co + 63) / 64));
        conv_bn_wmma<<<g, 256, 0, stream>>>(in, w,
            F(bn[1]) /*gamma*/, F(bn[0]) /*beta*/, F(bn[2]) /*mean*/, F(bn[3]) /*var*/,
            skip, outp, N, H, Wd, C, Co, R, S, st, pad, OH, OW, relu);
    };

    // ---- stem (padded C=4, S=8; (224+6-8)/2+1 == 112 still) + maxpool ----
    conv(P1, stemW, STEM_BN, nullptr, P0, 224, 224, 4, 64, 7, 8, 2, 3, 1);       // -> 112x112x64 (P0)
    maxpool3x3s2_nhwc<<<grid1((long long)N * 56 * 56 * 64), 256, 0, stream>>>(
        P0, P1, N, 64, 112, 112, 56, 56);                                        // -> 56x56x64 (P1)

    // ---- layer1 (56x56x64) ----
    conv(P1, bw1[0][0], bp[0][0].bn1, nullptr, P2, 56, 56, 64, 64, 3, 3, 1, 1, 1);
    conv(P2, bw2[0][0], bp[0][0].bn2, P1,      P0, 56, 56, 64, 64, 3, 3, 1, 1, 1); // act P0
    conv(P0, bw1[0][1], bp[0][1].bn1, nullptr, P2, 56, 56, 64, 64, 3, 3, 1, 1, 1);
    conv(P2, bw2[0][1], bp[0][1].bn2, P0,      P1, 56, 56, 64, 64, 3, 3, 1, 1, 1); // act P1

    // ---- layer2 (->28x28x128) ----
    conv(P1, bdw[1],    bp[1][0].dbn, nullptr, P2, 56, 56, 64, 128, 1, 1, 2, 0, 0); // skip path
    conv(P1, bw1[1][0], bp[1][0].bn1, nullptr, P0, 56, 56, 64, 128, 3, 3, 2, 1, 1);
    conv(P0, bw2[1][0], bp[1][0].bn2, P2,      P1, 28, 28, 128, 128, 3, 3, 1, 1, 1); // act P1
    conv(P1, bw1[1][1], bp[1][1].bn1, nullptr, P2, 28, 28, 128, 128, 3, 3, 1, 1, 1);
    conv(P2, bw2[1][1], bp[1][1].bn2, P1,      P0, 28, 28, 128, 128, 3, 3, 1, 1, 1); // act P0

    // ---- layer3 (->14x14x256) ----
    conv(P0, bdw[2],    bp[2][0].dbn, nullptr, P2, 28, 28, 128, 256, 1, 1, 2, 0, 0);
    conv(P0, bw1[2][0], bp[2][0].bn1, nullptr, P1, 28, 28, 128, 256, 3, 3, 2, 1, 1);
    conv(P1, bw2[2][0], bp[2][0].bn2, P2,      P0, 14, 14, 256, 256, 3, 3, 1, 1, 1); // act P0
    conv(P0, bw1[2][1], bp[2][1].bn1, nullptr, P1, 14, 14, 256, 256, 3, 3, 1, 1, 1);
    conv(P1, bw2[2][1], bp[2][1].bn2, P0,      P2, 14, 14, 256, 256, 3, 3, 1, 1, 1); // act P2

    // ---- layer4 (->7x7x512) ----
    conv(P2, bdw[3],    bp[3][0].dbn, nullptr, P1, 14, 14, 256, 512, 1, 1, 2, 0, 0);
    conv(P2, bw1[3][0], bp[3][0].bn1, nullptr, P0, 14, 14, 256, 512, 3, 3, 2, 1, 1);
    conv(P0, bw2[3][0], bp[3][0].bn2, P1,      P2, 7, 7, 512, 512, 3, 3, 1, 1, 1);   // act P2
    conv(P2, bw1[3][1], bp[3][1].bn1, nullptr, P0, 7, 7, 512, 512, 3, 3, 1, 1, 1);
    conv(P0, bw2[3][1], bp[3][1].bn2, P2,      P1, 7, 7, 512, 512, 3, 3, 1, 1, 1);   // act P1

    // ---- head ----
    avgpool7x7_nhwc<<<grid1((long long)N * 512), 256, 0, stream>>>(P1, pooled, N, 512);
    dim3 fg(1, 1024 / 32);
    gemm_wmma_f32out<<<fg, 256, 0, stream>>>(pooled, fcW, (float*)d_out, 64, 512, 1024, 1000);
}
